// MultiHeadedAttention_9560597201231
// MI455X (gfx1250) — matrix operand order repaired and checked by reading, not compiled, not run on hardware
//
#include <hip/hip_runtime.h>

// Problem constants (match reference)
#define BATCH 2
#define SEQ   2048
#define EMB   1024
#define HEADS 16
#define HDIM  64
#define NE3   (3 * EMB)
// 1/sqrt(EMB) * log2(e), folded into Q at QKV-store time so softmax uses exp2
#define Q_SCALE (0.03125f * 1.4426950408889634f)

typedef __attribute__((ext_vector_type(16))) __bf16 v16bf;
typedef __attribute__((ext_vector_type(8)))  __bf16 v8bf;
typedef __attribute__((ext_vector_type(4)))  __bf16 v4bf;
typedef __attribute__((ext_vector_type(2)))  __bf16 v2bf;
typedef __attribute__((ext_vector_type(8)))  float  v8f;
typedef __attribute__((ext_vector_type(4)))  unsigned int v4u;
typedef __attribute__((ext_vector_type(8)))  int v8i;
typedef __attribute__((ext_vector_type(4)))  int v4i;

static __device__ __forceinline__ v16bf make_frag(v8bf lo, v8bf hi) {
  return __builtin_shufflevector(lo, hi, 0,1,2,3,4,5,6,7,8,9,10,11,12,13,14,15);
}

// A-matrix fragment (16x32 bf16, ISA 7.12.2): lanes 0-15 hold row=lane with
// K {0..7} then {16..23}; lanes 16-31 hold row=lane-16 with K {8..15},{24..31}.
// Caller passes pointer to this lane's row start (row = lane&15).
static __device__ __forceinline__ v16bf load_frag_a(const __bf16* row_base, int lane) {
  const int ko = (lane >> 4) << 3;                 // 0 or 8
  v8bf lo = *(const v8bf*)(row_base + ko);         // K ko..ko+7
  v8bf hi = *(const v8bf*)(row_base + ko + 16);    // K ko+16..ko+23
  return make_frag(lo, hi);
}

// B-matrix fragment (32x16 bf16) read from a B^T (column-major-of-B) tile:
// lanes 0-15 hold col=lane with K {0..7},{16..23}; lanes 16-31 hold col=lane-16 with K {8..15},{24..31}.
// Caller passes pointer to this lane's column start (col = lane&15).
static __device__ __forceinline__ v16bf load_frag_b(const __bf16* col_base, int lane) {
  const int ko = (lane >> 4) << 3;                 // 0 or 8
  v8bf lo = *(const v8bf*)(col_base + ko);         // K ko..ko+7
  v8bf hi = *(const v8bf*)(col_base + ko + 16);    // K ko+16..ko+23
  return make_frag(lo, hi);
}

#define WMMA_BF16(a, b, c) \
  __builtin_amdgcn_wmma_f32_16x16x32_bf16(false, (a), false, (b), (short)0, (c), false, false)

// LDS byte offset of a __shared__ object (generic -> AS(3) addrspacecast).
static __device__ __forceinline__ unsigned lds_off_of(const void* p) {
  return (unsigned)(unsigned long long)(__attribute__((address_space(3))) const void*)p;
}

// Issue a 2D TDM load (Tensor Data Mover) of 2-byte elements into LDS.
//   tile = tile_d0 x tile_d1 elements; row stride in global = stride0 elements.
// D# per CDNA5 ISA: group0 {count, lds_addr, global_addr,
// type=2}; group1 {data_size=1(2B), tensor dims, tile dims, dim0 stride}.
// This toolchain exposes the 6-arg builtin (groups 2/3 + extra group + cpol).
static __device__ __forceinline__ void tdm_load_2d(
    unsigned lds_off, const void* gptr,
    unsigned tensor_d0, unsigned tensor_d1,
    unsigned tile_d0, unsigned tile_d1,
    unsigned long long stride0_elems) {
  unsigned long long ga = (unsigned long long)gptr;
  v4u g0 = { 1u,                                              // count=1, user D#
             lds_off,                                         // lds_addr [63:32]
             (unsigned)ga,                                    // global_addr lo
             (unsigned)((ga >> 32) & 0x1FFFFFFull) | (2u << 30) }; // addr hi | type=2
  v8i g1 = { (int)(1u << 16),                                 // data_size=1 -> 2 bytes
             (int)((tensor_d0 & 0xFFFFu) << 16),              // tensor_dim0 [79:48]
             (int)(((tensor_d0 >> 16) & 0xFFFFu) |
                   ((tensor_d1 & 0xFFFFu) << 16)),            // tensor_dim1 [111:80]
             (int)(((tensor_d1 >> 16) & 0xFFFFu) |
                   ((tile_d0 & 0xFFFFu) << 16)),              // tile_dim0 [127:112]
             (int)(tile_d1 & 0xFFFFu),                        // tile_dim1; tile_dim2=0
             (int)(unsigned)(stride0_elems & 0xFFFFFFFFull),  // dim0_stride lo
             (int)(unsigned)((stride0_elems >> 32) & 0xFFFFull),
             0 };
  v4i z4 = { 0, 0, 0, 0 };                                    // 2D: groups 2/3 unused
  v8i z8 = { 0, 0, 0, 0, 0, 0, 0, 0 };
  __builtin_amdgcn_tensor_load_to_lds(g0, g1, z4, z4, z8, 0);
}

// ---------------------------------------------------------------------------
// Kernel 1: qkv = hs @ w_qkv (fp32 in, bf16 out). Q pre-scaled by
// (1/sqrt(E))*log2(e) in [b,h,s,d]; K in [b,h,s,d]; V stored TRANSPOSED in
// [b,h,d,s] so kernel 2 gets both WMMA B^T operands as plain 2D tiles.
// Block tile 128x128, 8 waves (4Mx2N), wave 32x64, K-step 32.
// Double-buffered LDS: global loads for step t+1 are issued (into registers)
// before the WMMAs on step t, converted/stored to LDS after — load latency
// hides behind the matrix ops instead of serializing on s_wait_loadcnt 0.
// ---------------------------------------------------------------------------
__global__ __launch_bounds__(256)
void qkv_gemm_kernel(const float* __restrict__ hs, const float* __restrict__ w,
                     __bf16* __restrict__ qb, __bf16* __restrict__ kb,
                     __bf16* __restrict__ vtb) {
  __shared__ __bf16 lds_a [2][128 * 32];
  __shared__ __bf16 lds_bt[2][128 * 32];

  const int tid  = threadIdx.x;
  const int lane = tid & 31;
  const int wv   = tid >> 5;
  const int wm   = wv & 3;
  const int wn   = wv >> 2;
  const int m0   = blockIdx.y * 128;
  const int n0   = blockIdx.x * 128;
  const int nl   = lane & 15;
  const int half8 = (lane >> 4) << 3;

  // Per-thread staging coordinates (constant across K-steps).
  const int ar = tid >> 1,            ac4 = (tid & 1) << 4;   // A: 2 float4 pairs / row
  const int bk = tid >> 3,            bc4 = (tid & 7) << 4;   // B: row kk, 4-col group

  float4 fa[4], fb[4];
  // Issue all 8 global_load_b128 for K-step kt (no conversion yet).
  auto issue_loads = [&](int kt) {
    const float* abase = hs + (size_t)(m0 + ar) * EMB + kt * 32;
    fa[0] = *(const float4*)(abase + ac4 + 0);
    fa[1] = *(const float4*)(abase + ac4 + 4);
    fa[2] = *(const float4*)(abase + ac4 + 8);
    fa[3] = *(const float4*)(abase + ac4 + 12);
    const float* bbase = w + (size_t)(kt * 32 + bk) * NE3 + n0 + bc4;
    fb[0] = *(const float4*)(bbase + 0);
    fb[1] = *(const float4*)(bbase + 4);
    fb[2] = *(const float4*)(bbase + 8);
    fb[3] = *(const float4*)(bbase + 12);
  };
  // Convert fp32 -> bf16 and store staged registers into LDS buffer `buf`.
  auto store_tiles = [&](int buf) {
    #pragma unroll
    for (int i = 0; i < 4; ++i) {
      v4bf p = { (__bf16)fa[i].x, (__bf16)fa[i].y, (__bf16)fa[i].z, (__bf16)fa[i].w };
      *(v4bf*)(&lds_a[buf][ar * 32 + ac4 + i * 4]) = p;
    }
    #pragma unroll
    for (int i = 0; i < 4; ++i) {
      int c = bc4 + i * 4;
      lds_bt[buf][(c + 0) * 32 + bk] = (__bf16)fb[i].x;
      lds_bt[buf][(c + 1) * 32 + bk] = (__bf16)fb[i].y;
      lds_bt[buf][(c + 2) * 32 + bk] = (__bf16)fb[i].z;
      lds_bt[buf][(c + 3) * 32 + bk] = (__bf16)fb[i].w;
    }
  };

  v8f acc[8] = {};
  const int NT1 = EMB / 32;

  issue_loads(0);
  store_tiles(0);
  __syncthreads();

  for (int kt = 0; kt < NT1; ++kt) {
    const int cur = kt & 1;
    if (kt + 1 < NT1) issue_loads(kt + 1);   // loads in flight during WMMAs

    const __bf16* arow = lds_a[cur] + (wm * 32 + nl) * 32;
    v16bf a0 = load_frag_a(arow, lane);
    v16bf a1 = load_frag_a(arow + 16 * 32, lane);
    #pragma unroll
    for (int ni = 0; ni < 4; ++ni) {
      v16bf bfrag = load_frag_b(lds_bt[cur] + (wn * 64 + ni * 16 + nl) * 32, lane);
      acc[ni]     = WMMA_BF16(a0, bfrag, acc[ni]);
      acc[4 + ni] = WMMA_BF16(a1, bfrag, acc[4 + ni]);
    }

    if (kt + 1 < NT1) store_tiles(1 - cur);  // waits land here, after compute
    __syncthreads();
  }

  // Epilogue: whole block tile lands in exactly one of Q/K/V (128 | 1024).
  const int which = n0 >> 10;
  const float cscale = (which == 0) ? Q_SCALE : 1.0f;
  const int nseg0 = n0 & 1023;
  #pragma unroll
  for (int mi = 0; mi < 2; ++mi) {
    #pragma unroll
    for (int ni = 0; ni < 4; ++ni) {
      v8f c = acc[mi * 4 + ni];
      const int n = nseg0 + wn * 64 + ni * 16 + nl;   // col within segment
      const int h = n >> 6, d = n & 63;
      const int mbase = m0 + wm * 32 + mi * 16 + half8;
      const int b = mbase >> 11;                       // 128-row tile: same b for all r
      const size_t bh = (size_t)(b * HEADS + h);
      if (which == 2) {
        // V transposed [b,h,d,s]: consecutive r -> consecutive s; pack pairs.
        __bf16* vrow = vtb + (bh * HDIM + d) * SEQ + (mbase & (SEQ - 1));
        #pragma unroll
        for (int r = 0; r < 8; r += 2) {
          v2bf pk = { (__bf16)c[r], (__bf16)c[r + 1] };
          *(v2bf*)(vrow + r) = pk;
        }
      } else {
        __bf16* dst = (which == 0) ? qb : kb;
        #pragma unroll
        for (int r = 0; r < 8; ++r) {
          int s = (mbase + r) & (SEQ - 1);
          dst[((bh * SEQ + s) << 6) + d] = (__bf16)(c[r] * cscale);
        }
      }
    }
  }
}

// ---------------------------------------------------------------------------
// Kernel 2: flash attention over one (b,h). 8 waves x 16 queries = 128 q/block.
// K and V^T tiles are DMA'd into double-buffered LDS by the Tensor Data Mover
// (wave 0 issues; TENSORcnt is in-order so wait<=2 keeps next tile in flight),
// overlapping the DMA for tile t+1 with WMMA+softmax on tile t.
// ---------------------------------------------------------------------------
__global__ __launch_bounds__(256)
void attention_kernel(const __bf16* __restrict__ qb, const __bf16* __restrict__ kb,
                      const __bf16* __restrict__ vtb, float* __restrict__ out) {
  __shared__ __bf16 lds_k [2][32 * HDIM];     // [key][d]  (== B^T for Q·K^T)
  __shared__ __bf16 lds_vt[2][HDIM * 32];     // [d][key]  (== B^T for P·V)
  __shared__ __bf16 lds_p [8 * 16 * 32];      // per-wave P: [wave][row][key]

  const int tid  = threadIdx.x;
  const int lane = tid & 31;
  const int wv   = tid >> 5;
  const int nl   = lane & 15;
  const int half8 = (lane >> 4) << 3;
  const int bh = blockIdx.y;                  // b*HEADS + h
  const int q0 = blockIdx.x * 128 + wv * 16;

  const __bf16* Q  = qb  + (size_t)bh * SEQ * HDIM;
  const __bf16* K  = kb  + (size_t)bh * SEQ * HDIM;
  const __bf16* VT = vtb + (size_t)bh * HDIM * SEQ;   // [d][s]

  // Q fragments live in registers for the whole loop (A-layout, K-dim = 64).
  const __bf16* qrow = Q + (size_t)(q0 + nl) * HDIM;
  v16bf qa0 = load_frag_a(qrow, lane);
  v16bf qa1 = load_frag_a(qrow + 32, lane);

  v8f acc[4] = {};
  float mrow[8], lrow[8];
  #pragma unroll
  for (int r = 0; r < 8; ++r) { mrow[r] = -3.0e38f; lrow[r] = 0.0f; }

  const int NT = SEQ / 32;
  const unsigned koff0 = lds_off_of(lds_k[0]),  koff1 = lds_off_of(lds_k[1]);
  const unsigned voff0 = lds_off_of(lds_vt[0]), voff1 = lds_off_of(lds_vt[1]);

  if (wv == 0) {  // preload tile 0 into buffer 0
    tdm_load_2d(koff0, K,  HDIM, SEQ, HDIM, 32, HDIM);     // 32 keys x 64 d
    tdm_load_2d(voff0, VT, SEQ, HDIM, 32, HDIM, SEQ);      // 64 d x 32 keys
  }

  for (int kt = 0; kt < NT; ++kt) {
    const int cur = kt & 1;
    if (wv == 0) {
      if (kt + 1 < NT) {   // kick off next tile into the other buffer
        const __bf16* Kn  = K  + (size_t)(kt + 1) * 32 * HDIM;
        const __bf16* VTn = VT + (size_t)(kt + 1) * 32;
        tdm_load_2d(cur ? koff0 : koff1, Kn,  HDIM, SEQ, HDIM, 32, HDIM);
        tdm_load_2d(cur ? voff0 : voff1, VTn, SEQ, HDIM, 32, HDIM, SEQ);
        __builtin_amdgcn_s_wait_tensorcnt(2);   // current tile's pair done
      } else {
        __builtin_amdgcn_s_wait_tensorcnt(0);
      }
    }
    __syncthreads();

    const __bf16* lk = lds_k[cur];
    const __bf16* lv = lds_vt[cur];

    // Scores: (16 q x 64 d) · (64 d x 32 keys) -> two 16x16 f32 frags.
    v8f s0 = {}, s1 = {};
    {
      v16bf b;
      b = load_frag_b(lk + (0  + nl) * HDIM + 0,  lane); s0 = WMMA_BF16(qa0, b, s0);
      b = load_frag_b(lk + (0  + nl) * HDIM + 32, lane); s0 = WMMA_BF16(qa1, b, s0);
      b = load_frag_b(lk + (16 + nl) * HDIM + 0,  lane); s1 = WMMA_BF16(qa0, b, s1);
      b = load_frag_b(lk + (16 + nl) * HDIM + 32, lane); s1 = WMMA_BF16(qa1, b, s1);
    }

    // Online softmax (base 2; scale & log2e pre-folded into Q). Row m=r+half8
    // is shared by the 16 lanes of each half-wave; xor 1/2/4/8 reduces them.
    #pragma unroll
    for (int r = 0; r < 8; ++r) {
      float t = fmaxf(s0[r], s1[r]);
      t = fmaxf(t, __shfl_xor(t, 1)); t = fmaxf(t, __shfl_xor(t, 2));
      t = fmaxf(t, __shfl_xor(t, 4)); t = fmaxf(t, __shfl_xor(t, 8));
      float mnew  = fmaxf(mrow[r], t);
      float alpha = exp2f(mrow[r] - mnew);
      float p0 = exp2f(s0[r] - mnew);
      float p1 = exp2f(s1[r] - mnew);
      float rs = p0 + p1;
      rs += __shfl_xor(rs, 1); rs += __shfl_xor(rs, 2);
      rs += __shfl_xor(rs, 4); rs += __shfl_xor(rs, 8);
      lrow[r] = lrow[r] * alpha + rs;
      mrow[r] = mnew;
      // C-layout -> A-layout transpose of P through per-wave LDS.
      __bf16* prow = lds_p + (wv * 16 + r + half8) * 32;
      prow[nl]      = (__bf16)p0;
      prow[16 + nl] = (__bf16)p1;
      #pragma unroll
      for (int ni = 0; ni < 4; ++ni) acc[ni][r] *= alpha;
    }

    // acc += P (16x32) · V (32x64). Same-wave LDS ops are in-order.
    v16bf pa = load_frag_a(lds_p + (wv * 16 + nl) * 32, lane);
    #pragma unroll
    for (int ni = 0; ni < 4; ++ni) {
      v16bf b = load_frag_b(lv + (ni * 16 + nl) * 32, lane);
      acc[ni] = WMMA_BF16(pa, b, acc[ni]);
    }
    __syncthreads();   // release this buffer for the DMA two tiles ahead
  }

  // Epilogue: out[b][q][h*64+d] fp32.
  const int b = bh >> 4, h = bh & 15;
  float* obase = out + (size_t)b * SEQ * EMB + (size_t)h * HDIM;
  #pragma unroll
  for (int r = 0; r < 8; ++r) {
    float inv = 1.0f / lrow[r];
    float* op = obase + (size_t)(q0 + r + half8) * EMB;
    #pragma unroll
    for (int ni = 0; ni < 4; ++ni)
      op[ni * 16 + nl] = acc[ni][r] * inv;
  }
}

// ---------------------------------------------------------------------------
extern "C" void kernel_launch(void* const* d_in, const int* in_sizes, int n_in,
                              void* d_out, int out_size, void* d_ws, size_t ws_size,
                              hipStream_t stream) {
  const float* hs = (const float*)d_in[0];   // [2, 2048, 1024] fp32
  const float* w  = (const float*)d_in[1];   // [1024, 3072] fp32

  // Workspace: Q (pre-scaled) and K in [b,h,s,d]; V transposed in [b,h,d,s].
  const size_t per = (size_t)BATCH * HEADS * SEQ * HDIM;
  __bf16* qb  = (__bf16*)d_ws;
  __bf16* kb  = qb + per;
  __bf16* vtb = kb + per;

  dim3 g1(NE3 / 128, (BATCH * SEQ) / 128);   // (24, 32)
  qkv_gemm_kernel<<<g1, 256, 0, stream>>>(hs, w, qb, kb, vtb);

  dim3 g2(SEQ / 128, BATCH * HEADS);         // (16, 32)
  attention_kernel<<<g2, 256, 0, stream>>>(qb, kb, vtb, (float*)d_out);
}
